// AttnMLP_62234076119626
// MI455X (gfx1250) — matrix operand order repaired and checked by reading, not compiled, not run on hardware
//
#include <hip/hip_runtime.h>
#include <hip/hip_bf16.h>

#define D_EMBD 2048
#define NROWS  16384   // 8 * 2048 rows
#define NLAYER 4

typedef __bf16 bf16;
typedef __attribute__((ext_vector_type(16))) __bf16 v16bf;
typedef __attribute__((ext_vector_type(8)))  __bf16 v8bf;
typedef __attribute__((ext_vector_type(4)))  __bf16 v4bf;
typedef __attribute__((ext_vector_type(8)))  float  v8f;

__device__ __forceinline__ float new_gelu(float x) {
    const float c = 0.79788456080286535588f;  // sqrt(2/pi)
    float x3 = x * x * x;
    float t  = tanhf(c * (x + 0.044715f * x3));
    return 0.5f * x * (1.0f + t);
}

// ---- gfx1250 async global->LDS copy (ASYNCcnt-tracked DMA), 16B per lane ----
__device__ __forceinline__ void async_copy_b128(unsigned lds_off, const bf16* g) {
    // generic LDS address is {SHARED_BASE[63:32], offset[31:0]}; low 32 bits is
    // the LDS byte address the VDST operand expects.
    asm volatile("global_load_async_to_lds_b128 %0, %1, off"
                 :: "v"(lds_off), "v"(g)
                 : "memory");
}

__device__ __forceinline__ void wait_async0() {
#if __has_builtin(__builtin_amdgcn_s_wait_asynccnt)
    __builtin_amdgcn_s_wait_asynccnt(0);
#else
    asm volatile("s_wait_asynccnt 0x0" ::: "memory");
#endif
}

// ---------------- weight fp32 -> bf16 conversion ----------------
__global__ __launch_bounds__(256) void cvt_w_kernel(const float* __restrict__ W,
                                                    bf16* __restrict__ Wb) {
    size_t i = ((size_t)blockIdx.x * blockDim.x + threadIdx.x) * 4;
    float4 f = *(const float4*)(W + i);
    v4bf o;
    o[0] = (bf16)f.x; o[1] = (bf16)f.y; o[2] = (bf16)f.z; o[3] = (bf16)f.w;
    *(v4bf*)(Wb + i) = o;
}

// ---------------- LayerNorm (one row per block) -> bf16 ----------------
template <typename TIn>
__global__ __launch_bounds__(256) void ln_kernel(const TIn* __restrict__ x,
                                                 const float* __restrict__ gw,
                                                 const float* __restrict__ gb,
                                                 bf16* __restrict__ h) {
    const int row = blockIdx.x;
    const TIn* xr = x + (size_t)row * D_EMBD;
    bf16* hr      = h + (size_t)row * D_EMBD;

    const int base = threadIdx.x * 8;
    float v[8];
    float s = 0.0f, q = 0.0f;
#pragma unroll
    for (int j = 0; j < 8; ++j) {
        float f = (float)xr[base + j];
        v[j] = f;
        s += f;
        q += f * f;
    }
    // wave32 reduce
#pragma unroll
    for (int off = 16; off > 0; off >>= 1) {
        s += __shfl_xor(s, off, 32);
        q += __shfl_xor(q, off, 32);
    }
    __shared__ float ss[8], sq[8];
    const int wid = threadIdx.x >> 5, lane = threadIdx.x & 31;
    if (lane == 0) { ss[wid] = s; sq[wid] = q; }
    __syncthreads();
    s = 0.0f; q = 0.0f;
#pragma unroll
    for (int w = 0; w < 8; ++w) { s += ss[w]; q += sq[w]; }

    const float mu   = s * (1.0f / (float)D_EMBD);
    const float var  = q * (1.0f / (float)D_EMBD) - mu * mu;
    const float rstd = rsqrtf(var + 1e-5f);
#pragma unroll
    for (int j = 0; j < 8; ++j) {
        float nv = (v[j] - mu) * rstd * gw[base + j] + gb[base + j];
        hr[base + j] = (bf16)nv;
    }
}

// ---------------- bf16 WMMA GEMM: out[m,n] = gelu(A[m,:] . Bw[n,:] + bias[n]) ----
// A: [NROWS, 2048] bf16 row-major (LN output)
// Bw: [2048, 2048] bf16 row-major, rows are output features (torch Linear layout)
union FragAB { v16bf v; v8bf h[2]; };

template <typename TOut>
__global__ __launch_bounds__(256) void gemm_kernel(const bf16* __restrict__ A,
                                                   const bf16* __restrict__ Bw,
                                                   const float* __restrict__ bias,
                                                   TOut* __restrict__ out) {
    constexpr int BM = 128, BN = 128, BK = 32, LDK = BK + 8;  // pad to dodge banks
    constexpr int KD = D_EMBD;
    __shared__ bf16 sA[2][BM * LDK];
    __shared__ bf16 sB[2][BN * LDK];

    const int tid  = threadIdx.x;
    const int lane = tid & 31;
    const int wave = tid >> 5;
    const int lrow = lane & 15;   // selects M (A) / N (B) / N (C)
    const int lhi  = lane >> 4;   // selects K-half (A,B) / M-half (C)
    const int wm   = (wave & 1) * 64;   // 2 waves along M
    const int wn   = (wave >> 1) * 32;  // 4 waves along N

    const size_t m0 = (size_t)blockIdx.y * BM;
    const size_t n0 = (size_t)blockIdx.x * BN;
    const bf16* Ag = A  + m0 * KD;
    const bf16* Bg = Bw + n0 * KD;

    // global->LDS coordinates: 256 threads, each 2x16B per matrix per tile
    const int lr = tid >> 2;        // 0..63  (row within tile, +64 on 2nd pass)
    const int lc = (tid & 3) * 8;   // 0,8,16,24 (k element offset)

    v8f acc[4][2] = {};

    // async-DMA one k-tile (A+B) into LDS buffer `buf`
    auto issueTile = [&](int buf, int kt) {
        const int k0 = kt * BK;
#pragma unroll
        for (int t = 0; t < 2; ++t) {
            int r = lr + t * 64;
            async_copy_b128((unsigned)(size_t)&sA[buf][r * LDK + lc],
                            Ag + (size_t)r * KD + k0 + lc);
            async_copy_b128((unsigned)(size_t)&sB[buf][r * LDK + lc],
                            Bg + (size_t)r * KD + k0 + lc);
        }
    };

    issueTile(0, 0);

    const int kTiles = KD / BK;  // 64
    for (int kt = 0; kt < kTiles; ++kt) {
        const int  cur  = kt & 1;
        const bool more = (kt + 1) < kTiles;

        wait_async0();      // this wave's tile-kt copies have landed in LDS
        __syncthreads();    // all waves' copies landed; everyone done reading buf cur^1

        if (more)           // DMA tile kt+1 into the other buffer, overlaps WMMAs
            issueTile(cur ^ 1, kt + 1);
        if (kt + 2 < kTiles) {  // gfx1250 global_prefetch_b8 for tile after next
            const int k0 = (kt + 2) * BK;
            __builtin_prefetch(Ag + (size_t)lr * KD + k0 + lc, 0, 1);
            __builtin_prefetch(Bg + (size_t)lr * KD + k0 + lc, 0, 1);
        }

        // A frags: 16x32 bf16; lane<16: K {0..7,16..23}, lane>=16: K {8..15,24..31}
        FragAB af[4];
#pragma unroll
        for (int mi = 0; mi < 4; ++mi) {
            const bf16* p = &sA[cur][(wm + mi * 16 + lrow) * LDK];
            af[mi].h[0] = *(const v8bf*)(p + 8 * lhi);
            af[mi].h[1] = *(const v8bf*)(p + 16 + 8 * lhi);
        }
        // B frags: 32x16 bf16, column N=lrow per lane; lane<16: K {0..7,16..23}, else K {8..15,24..31}
        FragAB bfr[2];
#pragma unroll
        for (int ni = 0; ni < 2; ++ni) {
            const bf16* p = &sB[cur][(wn + ni * 16 + lrow) * LDK + 8 * lhi];
            bfr[ni].h[0] = *(const v8bf*)(p);
            bfr[ni].h[1] = *(const v8bf*)(p + 16);
        }

#pragma unroll
        for (int mi = 0; mi < 4; ++mi)
#pragma unroll
            for (int ni = 0; ni < 2; ++ni)
                acc[mi][ni] = __builtin_amdgcn_wmma_f32_16x16x32_bf16(
                    false, af[mi].v, false, bfr[ni].v, (short)0, acc[mi][ni],
                    false, false);
    }

    // epilogue: bias + GELU; C/D layout: VGPR r -> M = r + 8*lhi, lanes -> N
#pragma unroll
    for (int ni = 0; ni < 2; ++ni) {
        const int   n  = (int)n0 + wn + ni * 16 + lrow;
        const float bv = bias[n];
#pragma unroll
        for (int mi = 0; mi < 4; ++mi) {
#pragma unroll
            for (int r = 0; r < 8; ++r) {
                const size_t m = m0 + wm + mi * 16 + r + 8 * lhi;
                float vv = acc[mi][ni][r] + bv;
                out[m * (size_t)D_EMBD + n] = (TOut)new_gelu(vv);
            }
        }
    }
}

// ---------------- host ----------------
extern "C" void kernel_launch(void* const* d_in, const int* in_sizes, int n_in,
                              void* d_out, int out_size, void* d_ws, size_t ws_size,
                              hipStream_t stream) {
    const float* x   = (const float*)d_in[0];  // [8,2048,2048]
    const float* W   = (const float*)d_in[1];  // [4,2048,2048] (out,in)
    const float* b   = (const float*)d_in[2];  // [4,2048]
    const float* lnw = (const float*)d_in[3];  // [4,2048]
    const float* lnb = (const float*)d_in[4];  // [4,2048]
    float* out = (float*)d_out;

    char* ws = (char*)d_ws;
    const size_t wbBytes = (size_t)NLAYER * D_EMBD * D_EMBD * sizeof(bf16);  // 32 MiB
    const size_t hBytes  = (size_t)NROWS * D_EMBD * sizeof(bf16);            // 64 MiB
    bf16* Wb = (bf16*)ws;                        // bf16 weights, all layers
    bf16* h  = (bf16*)(ws + wbBytes);            // LN output (bf16)
    bf16* xt = (bf16*)(ws + wbBytes + hBytes);   // inter-layer activations (bf16)

    {   // convert all weights once per launch (deterministic, no caching)
        int nvec = (NLAYER * D_EMBD * D_EMBD) / 4;  // float4 per thread
        cvt_w_kernel<<<nvec / 256, 256, 0, stream>>>(W, Wb);
    }

    dim3 gGrid(D_EMBD / 128, NROWS / 128);  // (16, 128)
    for (int i = 0; i < NLAYER; ++i) {
        const float* gw = lnw + (size_t)i * D_EMBD;
        const float* gb = lnb + (size_t)i * D_EMBD;
        const bf16*  Wi = Wb + (size_t)i * D_EMBD * D_EMBD;
        const float* bi = b + (size_t)i * D_EMBD;

        if (i == 0)
            ln_kernel<float><<<NROWS, 256, 0, stream>>>(x, gw, gb, h);
        else
            ln_kernel<bf16><<<NROWS, 256, 0, stream>>>(xt, gw, gb, h);

        if (i == NLAYER - 1)
            gemm_kernel<float><<<gGrid, 256, 0, stream>>>(h, Wi, bi, out);
        else
            gemm_kernel<bf16><<<gGrid, 256, 0, stream>>>(h, Wi, bi, xt);
    }
    (void)in_sizes; (void)n_in; (void)out_size; (void)ws_size;
}
